// MultiHeadLatentAttention_82197084110873
// MI455X (gfx1250) — matrix operand order repaired and checked by reading, not compiled, not run on hardware
//
#include <hip/hip_runtime.h>
#include <hip/hip_bf16.h>
#include <math.h>

// ---------------------------------------------------------------------------
// Types / helpers
// ---------------------------------------------------------------------------
typedef __bf16 bf16_t;
typedef bf16_t v16bf __attribute__((ext_vector_type(16)));
typedef float  v8f   __attribute__((ext_vector_type(8)));
typedef int    v4i_t __attribute__((vector_size(16)));   // 16B chunk for async DMA

#define WMMA_BF16(A, B, C) \
  __builtin_amdgcn_wmma_f32_16x16x32_bf16(false, (A), false, (B), (short)0, (C), false, false)

__device__ __forceinline__ bf16_t f2bf(float f) {
  union { float f; unsigned u; } x;
  x.f = f;
  unsigned r = x.u + 0x7fffu + ((x.u >> 16) & 1u);   // round-to-nearest-even
  unsigned short h = (unsigned short)(r >> 16);
  bf16_t o;
  __builtin_memcpy(&o, &h, 2);
  return o;
}

__device__ __forceinline__ v8f v8f_zero() {
  v8f z;
#pragma unroll
  for (int i = 0; i < 8; ++i) z[i] = 0.0f;
  return z;
}

// A-fragment (16x32 bf16): lane = row; element e comes from
// k = 16*half + e (one contiguous 32B chunk per lane).
__device__ __forceinline__ v16bf load_afrag(const bf16_t* rowp, int half) {
  v16bf a;
#pragma unroll
  for (int e = 0; e < 16; ++e) a[e] = rowp[half * 16 + e];
  return a;
}

// B-fragment (32x16 bf16, same k order as A-fragment): lane = column; K = 16*half + e,
// one contiguous 32B chunk per lane.
__device__ __forceinline__ v16bf load_bfrag(const bf16_t* colp, int half) {
  v16bf b;
#pragma unroll
  for (int e = 0; e < 16; ++e) b[e] = colp[half * 16 + e];
  return b;
}

// ---------------------------------------------------------------------------
// Async global->LDS (CDNA5 GLOBAL_LOAD_ASYNC_TO_LDS_B128, ASYNCcnt-tracked).
// Builtin signature (from hipcc diagnostic): (v4i AS1*, v4i AS3*, Ii, Ii).
// Guarded: falls back to a plain 16B copy if the builtin is unavailable.
// ---------------------------------------------------------------------------
#if __has_builtin(__builtin_amdgcn_global_load_async_to_lds_b128)
#define MLA_HAS_ASYNC_LDS 1
#else
#define MLA_HAS_ASYNC_LDS 0
#endif

__device__ __forceinline__ void copy16_g2l(const bf16_t* g, bf16_t* l) {
#if MLA_HAS_ASYNC_LDS
  __builtin_amdgcn_global_load_async_to_lds_b128(
      (__attribute__((address_space(1))) v4i_t*)(void*)const_cast<bf16_t*>(g),
      (__attribute__((address_space(3))) v4i_t*)(void*)l, 0, 0);
#else
  *(uint4*)l = *(const uint4*)g;
#endif
}

__device__ __forceinline__ void async_wait_all() {
#if MLA_HAS_ASYNC_LDS
#if __has_builtin(__builtin_amdgcn_s_wait_asynccnt)
  __builtin_amdgcn_s_wait_asynccnt(0);
#else
  asm volatile("s_wait_asynccnt 0x0" ::: "memory");
#endif
#endif
}

// Problem constants
constexpr int Bc  = 2;
constexpr int Sc_ = 2048;
constexpr int Hc  = 16;
constexpr int DHc = 128;
constexpr int DRc = 64;
constexpr int Mtot = Bc * Sc_;   // 4096 token rows

// ---------------------------------------------------------------------------
// fp32 -> bf16 bulk conversion (one-shot; removes converts from GEMM loops)
// ---------------------------------------------------------------------------
__global__ __launch_bounds__(256) void cvt_f32_bf16_kernel(
    const float* __restrict__ in, bf16_t* __restrict__ out, int n)
{
  int i = (blockIdx.x * blockDim.x + threadIdx.x) * 4;
  if (i + 3 < n) {
    const float4 v = *(const float4*)(in + i);
    out[i]     = f2bf(v.x);
    out[i + 1] = f2bf(v.y);
    out[i + 2] = f2bf(v.z);
    out[i + 3] = f2bf(v.w);
  } else {
    for (; i < n; ++i) out[i] = f2bf(in[i]);
  }
}

// ---------------------------------------------------------------------------
// bf16-WMMA GEMM:  C[M,N] = A[M,K] @ W[N,K]^T + bias[N]
// A,W bf16; C fp32 or bf16. 256 threads = 8 waves, block tile 128x128,
// K staged 32 at a time through DOUBLE-BUFFERED LDS filled by the async
// global->LDS DMA (overlapped with WMMA on the current buffer).
// ---------------------------------------------------------------------------
template <typename OutT>
__global__ __launch_bounds__(256) void gemm_bias_kernel(
    const bf16_t* __restrict__ A, const bf16_t* __restrict__ W,
    const float* __restrict__ bias, OutT* __restrict__ C,
    int M, int N, int K)
{
  __shared__ bf16_t As[2][128 * 32];
  __shared__ bf16_t Ws[2][128 * 32];

  const int t     = threadIdx.x;
  const int lane  = t & 31;
  const int wave  = t >> 5;
  const int row   = lane & 15;
  const int half  = lane >> 4;
  const int waveM = wave & 3;       // 4 x 2 wave grid: 32-row x 64-col per wave
  const int waveN = wave >> 2;
  const int bm = blockIdx.y * 128;
  const int bn = blockIdx.x * 128;

  // Stage one 128x32 K-slab of A and W into LDS buffer `buf` (16B chunks).
  auto stage = [&](int buf, int kb) {
    for (int ch = t; ch < 512; ch += 256) {     // 512 chunks of 8 bf16
      const int r  = ch >> 2;
      const int c8 = (ch & 3) * 8;
      copy16_g2l(&A[(size_t)(bm + r) * K + kb + c8], &As[buf][r * 32 + c8]);
      copy16_g2l(&W[(size_t)(bn + r) * K + kb + c8], &Ws[buf][r * 32 + c8]);
    }
  };

  v8f acc[2][4];
#pragma unroll
  for (int i = 0; i < 2; ++i)
#pragma unroll
    for (int j = 0; j < 4; ++j) acc[i][j] = v8f_zero();

  const int nk = K >> 5;
  stage(0, 0);
  async_wait_all();
  __syncthreads();

  for (int ks = 0; ks < nk; ++ks) {
    const int cur = ks & 1;
    if (ks + 1 < nk) stage(cur ^ 1, (ks + 1) << 5);   // DMA next slab

    v16bf Af[2], Bf[4];
#pragma unroll
    for (int i = 0; i < 2; ++i)
      Af[i] = load_afrag(&As[cur][(waveM * 32 + i * 16 + row) * 32], half);
#pragma unroll
    for (int j = 0; j < 4; ++j)
      Bf[j] = load_bfrag(&Ws[cur][(waveN * 64 + j * 16 + row) * 32], half);
#pragma unroll
    for (int i = 0; i < 2; ++i)
#pragma unroll
      for (int j = 0; j < 4; ++j)
        acc[i][j] = WMMA_BF16(Af[i], Bf[j], acc[i][j]);

    async_wait_all();     // next slab landed (DMA overlapped with WMMAs above)
    __syncthreads();
  }

  // Epilogue: bias + store (C layout: VGPR r -> M = r + 8*half, N = lane&15)
#pragma unroll
  for (int j = 0; j < 4; ++j) {
    const int n = bn + waveN * 64 + j * 16 + row;
    const float bv = bias[n];
#pragma unroll
    for (int i = 0; i < 2; ++i) {
#pragma unroll
      for (int r = 0; r < 8; ++r) {
        const int m = bm + waveM * 32 + i * 16 + r + half * 8;
        const float val = acc[i][j][r] + bv;
        if constexpr (sizeof(OutT) == 2)
          C[(size_t)m * N + n] = (OutT)f2bf(val);
        else
          C[(size_t)m * N + n] = (OutT)val;
      }
    }
  }
}

// ---------------------------------------------------------------------------
// Rotate-half RoPE, in place on a [M, H*64] bf16 buffer. One thread per pair.
// ---------------------------------------------------------------------------
__global__ __launch_bounds__(256) void rope_kernel(
    bf16_t* __restrict__ t, int M, int S, int H)
{
  const int idx = blockIdx.x * blockDim.x + threadIdx.x;
  const int i  = idx & 31;        // pair index 0..31
  const int hm = idx >> 5;
  const int h  = hm % H;
  const int m  = hm / H;
  if (m >= M) return;
  const int s = m % S;
  const float inv = __expf(-(float)i * 0.28782313662425572f);  // 10000^{-i/32}
  const float ang = (float)s * inv;
  float sn, cs;
  __sincosf(ang, &sn, &cs);
  const size_t base = (size_t)m * (H * 64) + h * 64 + i;
  const float t1 = (float)t[base], t2 = (float)t[base + 32];
  t[base]      = f2bf(t1 * cs - t2 * sn);
  t[base + 32] = f2bf(t2 * cs + t1 * sn);
}

// ---------------------------------------------------------------------------
// Streaming causal flash attention, block-cooperative:
// one block = one (b,h) x 64 query rows (4 waves x 16-row q-tiles).
// Each 32-key chunk (K: 192 dims, V: 128 dims) is staged ONCE per block into
// double-buffered LDS by the async global->LDS DMA, overlapped with the
// 20 WMMAs/chunk on the previous buffer. All waves run the block's full
// causal range; per-wave masking keeps the online softmax exact.
// LDS: 2*(32x192) K + 2*(32x128) V + 4*(16x32) probs = 44 KB.
// ---------------------------------------------------------------------------
__global__ __launch_bounds__(128) void mla_attn_kernel(
    const bf16_t* __restrict__ qc, const bf16_t* __restrict__ qr,
    const bf16_t* __restrict__ kcn, const bf16_t* __restrict__ kr,
    const bf16_t* __restrict__ vv, bf16_t* __restrict__ out)
{
  __shared__ bf16_t Ks[2][32 * 192];    // [key][content 0..127 | rope 128..191]
  __shared__ bf16_t Vs[2][32 * 128];    // [key][dim]
  __shared__ bf16_t plds[4][16 * 32];   // per-wave prob staging

  const int t    = threadIdx.x;
  const int lane = t & 31;
  const int wave = t >> 5;
  const int qg = blockIdx.x & 31;              // 64-query group
  const int h  = (blockIdx.x >> 5) & (Hc - 1);
  const int b  = blockIdx.x >> 9;
  const int row  = lane & 15;
  const int half = lane >> 4;
  const int ldq = Hc * DHc;   // 2048
  const int ldr = Hc * DRc;   // 1024
  const int q0 = qg * 64 + wave * 16;          // this wave's query tile
  const int mrow = b * Sc_ + q0 + row;

  // Stage one 32-key chunk (K content + K rope + V) into LDS buffer `buf`.
  auto stage = [&](int buf, int k0) {
    const size_t kcBase = (size_t)(b * Sc_ + k0) * ldq + h * DHc;
    const size_t krBase = (size_t)(b * Sc_ + k0) * ldr + h * DRc;
    for (int c = t; c < 512; c += 128) {        // K content: 32 x 128
      const int key = c >> 4, c8 = (c & 15) * 8;
      copy16_g2l(kcn + kcBase + (size_t)key * ldq + c8, &Ks[buf][key * 192 + c8]);
    }
    for (int c = t; c < 256; c += 128) {        // K rope: 32 x 64
      const int key = c >> 3, c8 = (c & 7) * 8;
      copy16_g2l(kr + krBase + (size_t)key * ldr + c8, &Ks[buf][key * 192 + 128 + c8]);
    }
    for (int c = t; c < 512; c += 128) {        // V: 32 x 128
      const int key = c >> 4, c8 = (c & 15) * 8;
      copy16_g2l(vv + kcBase + (size_t)key * ldq + c8, &Vs[buf][key * 128 + c8]);
    }
  };

  // Q fragments: 6 x (16x32 bf16); 4 from q_cnt, 2 from roped q_r.
  v16bf Qf[6];
#pragma unroll
  for (int f = 0; f < 6; ++f) {
    const bf16_t* src = (f < 4)
        ? (qc + (size_t)mrow * ldq + h * DHc + f * 32)
        : (qr + (size_t)mrow * ldr + h * DRc + (f - 4) * 32);
    Qf[f] = load_afrag(src, half);
  }

  v8f O[8];
#pragma unroll
  for (int c = 0; c < 8; ++c) O[c] = v8f_zero();
  float mrun[8], lrun[8];
#pragma unroll
  for (int r = 0; r < 8; ++r) { mrun[r] = -1e30f; lrun[r] = 0.0f; }

  const float scale = 0.07216878364870323f;   // 1/sqrt(192)
  const int nCh = qg * 2 + 2;                 // chunks covering block's causal range

  stage(0, 0);
  async_wait_all();
  __syncthreads();

  for (int kch = 0; kch < nCh; ++kch) {
    const int cur = kch & 1;
    if (kch + 1 < nCh) stage(cur ^ 1, (kch + 1) * 32);   // DMA next chunk
    const int k0 = kch * 32;

    // --- scores: two 16x16 tiles over d = 192, K fragments from LDS ---
    v8f Sc2[2];
#pragma unroll
    for (int tile = 0; tile < 2; ++tile) {
      const bf16_t* krow = &Ks[cur][(tile * 16 + row) * 192];  // lane = key column
      v8f acc = v8f_zero();
#pragma unroll
      for (int f = 0; f < 6; ++f)
        acc = WMMA_BF16(Qf[f], load_bfrag(krow + f * 32, half), acc);
      Sc2[tile] = acc;
    }

    // --- online softmax (C layout: row = r + 8*half, col = lane&15) ---
    const bool needMask = (k0 + 31 > q0);
#pragma unroll
    for (int r = 0; r < 8; ++r) {
      float a  = Sc2[0][r] * scale;
      float bb = Sc2[1][r] * scale;
      if (needMask) {
        const int qrow = q0 + r + half * 8;
        if (k0 + row > qrow)      a  = -1e9f;
        if (k0 + 16 + row > qrow) bb = -1e9f;
      }
      float mx = fmaxf(a, bb);
      mx = fmaxf(mx, __shfl_xor(mx, 1, 32));
      mx = fmaxf(mx, __shfl_xor(mx, 2, 32));
      mx = fmaxf(mx, __shfl_xor(mx, 4, 32));
      mx = fmaxf(mx, __shfl_xor(mx, 8, 32));
      const float mnew = fmaxf(mrun[r], mx);
      const float pa = __expf(a - mnew);
      const float pb = __expf(bb - mnew);
      float sum = pa + pb;
      sum += __shfl_xor(sum, 1, 32);
      sum += __shfl_xor(sum, 2, 32);
      sum += __shfl_xor(sum, 4, 32);
      sum += __shfl_xor(sum, 8, 32);
      const float alpha = __expf(mrun[r] - mnew);
      lrun[r] = lrun[r] * alpha + sum;
      mrun[r] = mnew;
#pragma unroll
      for (int c = 0; c < 8; ++c) O[c][r] *= alpha;
      plds[wave][(r + half * 8) * 32 + row]      = f2bf(pa);
      plds[wave][(r + half * 8) * 32 + 16 + row] = f2bf(pb);
    }

    // --- repack probs C-layout -> A-layout (same-wave LDS, in-order) ---
    const v16bf Pf = load_afrag(&plds[wave][row * 32], half);

    // --- P @ V : 8 WMMAs, V fragments from LDS (lane = dim column) ---
#pragma unroll
    for (int c = 0; c < 8; ++c) {
      const bf16_t* vsrc = &Vs[cur][(half * 16) * 128 + c * 16 + row];
      v16bf vf;
#pragma unroll
      for (int e = 0; e < 16; ++e) vf[e] = vsrc[e * 128];
      O[c] = WMMA_BF16(Pf, vf, O[c]);
    }

    async_wait_all();     // next chunk landed (DMA overlapped with WMMAs above)
    __syncthreads();
  }

  // --- normalize and store bf16 [M, H*128] ---
  float invl[8];
#pragma unroll
  for (int r = 0; r < 8; ++r) invl[r] = 1.0f / lrun[r];
#pragma unroll
  for (int c = 0; c < 8; ++c) {
#pragma unroll
    for (int r = 0; r < 8; ++r) {
      const int m = b * Sc_ + q0 + r + half * 8;
      out[(size_t)m * ldq + h * DHc + c * 16 + row] = f2bf(O[c][r] * invl[r]);
    }
  }
}

// ---------------------------------------------------------------------------
// Host launcher
// ---------------------------------------------------------------------------
extern "C" void kernel_launch(void* const* d_in, const int* in_sizes, int n_in,
                              void* d_out, int out_size, void* d_ws, size_t ws_size,
                              hipStream_t stream)
{
  (void)in_sizes; (void)n_in; (void)out_size;
  const float* x   = (const float*)d_in[0];
  const float* Wkd = (const float*)d_in[1];  const float* bkd = (const float*)d_in[2];
  const float* Wqd = (const float*)d_in[3];  const float* bqd = (const float*)d_in[4];
  const float* Wku = (const float*)d_in[5];  const float* bku = (const float*)d_in[6];
  const float* Wvu = (const float*)d_in[7];  const float* bvu = (const float*)d_in[8];
  const float* Wqu = (const float*)d_in[9];  const float* bqu = (const float*)d_in[10];
  const float* Wkr = (const float*)d_in[11]; const float* bkr = (const float*)d_in[12];
  const float* Wqr = (const float*)d_in[13]; const float* bqr = (const float*)d_in[14];
  const float* Wo  = (const float*)d_in[15]; const float* bo  = (const float*)d_in[16];
  float* out = (float*)d_out;

  bf16_t* ws = (bf16_t*)d_ws;
  size_t off = 0;
  auto carve = [&](size_t n) { bf16_t* p = ws + off; off += n; return p; };
  // bf16 copies of input / weights
  bf16_t* xb   = carve((size_t)Mtot * 2048);
  bf16_t* Wkdb = carve((size_t)512  * 2048);
  bf16_t* Wqdb = carve((size_t)1536 * 2048);
  bf16_t* Wkub = carve((size_t)2048 * 512);
  bf16_t* Wvub = carve((size_t)2048 * 512);
  bf16_t* Wqub = carve((size_t)2048 * 1536);
  bf16_t* Wkrb = carve((size_t)1024 * 2048);
  bf16_t* Wqrb = carve((size_t)1024 * 1536);
  bf16_t* Wob  = carve((size_t)2048 * 2048);
  // bf16 intermediates
  bf16_t* kv_c  = carve((size_t)Mtot * 512);
  bf16_t* q_c   = carve((size_t)Mtot * 1536);
  bf16_t* k_cnt = carve((size_t)Mtot * 2048);
  bf16_t* vbuf  = carve((size_t)Mtot * 2048);
  bf16_t* q_cnt = carve((size_t)Mtot * 2048);
  bf16_t* k_r   = carve((size_t)Mtot * 1024);
  bf16_t* q_r   = carve((size_t)Mtot * 1024);
  bf16_t* attn  = carve((size_t)Mtot * 2048);
  if (ws_size < off * sizeof(bf16_t)) return;

  auto cvt = [&](const float* in, bf16_t* o, size_t n) {
    hipLaunchKernelGGL(cvt_f32_bf16_kernel, dim3((unsigned)(n / 1024)), dim3(256),
                       0, stream, in, o, (int)n);
  };
  cvt(x,   xb,   (size_t)Mtot * 2048);
  cvt(Wkd, Wkdb, (size_t)512  * 2048);
  cvt(Wqd, Wqdb, (size_t)1536 * 2048);
  cvt(Wku, Wkub, (size_t)2048 * 512);
  cvt(Wvu, Wvub, (size_t)2048 * 512);
  cvt(Wqu, Wqub, (size_t)2048 * 1536);
  cvt(Wkr, Wkrb, (size_t)1024 * 2048);
  cvt(Wqr, Wqrb, (size_t)1024 * 1536);
  cvt(Wo,  Wob,  (size_t)2048 * 2048);

  const dim3 blk(256);
  auto gemm_bf = [&](const bf16_t* A, const bf16_t* W, const float* bias,
                     bf16_t* C, int M, int N, int K) {
    dim3 grid(N / 128, M / 128);
    hipLaunchKernelGGL((gemm_bias_kernel<bf16_t>), grid, blk, 0, stream,
                       A, W, bias, C, M, N, K);
  };

  gemm_bf(xb,   Wkdb, bkd, kv_c,  Mtot, 512,  2048);   // kv latent
  gemm_bf(xb,   Wqdb, bqd, q_c,   Mtot, 1536, 2048);   // q latent
  gemm_bf(kv_c, Wkub, bku, k_cnt, Mtot, 2048, 512);    // key content
  gemm_bf(kv_c, Wvub, bvu, vbuf,  Mtot, 2048, 512);    // value
  gemm_bf(q_c,  Wqub, bqu, q_cnt, Mtot, 2048, 1536);   // query content
  gemm_bf(xb,   Wkrb, bkr, k_r,   Mtot, 1024, 2048);   // key rope feats
  gemm_bf(q_c,  Wqrb, bqr, q_r,   Mtot, 1024, 1536);   // query rope feats

  const int ropeTotal = Mtot * Hc * 32;
  hipLaunchKernelGGL(rope_kernel, dim3(ropeTotal / 256), dim3(256), 0, stream,
                     k_r, Mtot, Sc_, Hc);
  hipLaunchKernelGGL(rope_kernel, dim3(ropeTotal / 256), dim3(256), 0, stream,
                     q_r, Mtot, Sc_, Hc);

  // one block per (b, h, 64-query group): 2*16*32 = 1024 blocks
  hipLaunchKernelGGL(mla_attn_kernel, dim3(Bc * Hc * (Sc_ / 64)), dim3(128),
                     0, stream, q_cnt, q_r, k_cnt, k_r, vbuf, attn);

  // output projection -> fp32
  {
    dim3 grid(2048 / 128, Mtot / 128);
    hipLaunchKernelGGL((gemm_bias_kernel<float>), grid, blk, 0, stream,
                       attn, Wob, bo, out, Mtot, 2048, 2048);
  }
}
